// DINO_Iterative_StereoNet_4277787427101
// MI455X (gfx1250) — matrix operand order repaired and checked by reading, not compiled, not run on hardware
//
#include <hip/hip_runtime.h>

// ---------------------------------------------------------------------------
// DINO iterative stereo net on MI455X (gfx1250, wave32, WMMA).
// Convs: implicit GEMM on v_wmma_f32_16x16x32_bf16 (f32 accumulate).
//  - weights packed [CO_pad][tap(ky,kx)][CI_pad]  (no div in hot loop)
//  - bf16 activations NHWC with per-pixel stride CI_pad -> one contiguous
//    32B load per staged chunk; async GLOBAL_LOAD_ASYNC_TO_LDS_B128 (ASYNCcnt)
//    with plain b128 load/store fallback.
//  - WG = 4 waves, 64 co x 64 px, 16 WMMAs per K chunk.
// Assumes ws_size >= ~210 MB.
// ---------------------------------------------------------------------------

typedef __attribute__((ext_vector_type(16))) __bf16 v16bf;
typedef __attribute__((ext_vector_type(8)))  __bf16 v8bf;
typedef __attribute__((ext_vector_type(8)))  float  v8f;

#define HN 56
#define WN 112
#define HWN 6272
#define BN 2

#if defined(__AMDGCN__) && __has_builtin(__builtin_amdgcn_global_load_async_to_lds_b128) && __has_builtin(__builtin_amdgcn_s_wait_asynccnt)
#define HAVE_ASYNC_LDS 1
typedef __attribute__((ext_vector_type(4))) int v4ia;
typedef __attribute__((address_space(1))) v4ia* gas4_t;   // global <4 x i32>*
typedef __attribute__((address_space(3))) v4ia* las4_t;   // LDS    <4 x i32>*
#else
#define HAVE_ASYNC_LDS 0
#endif

static inline int cdiv_h(int a, int b) { return (a + b - 1) / b; }

#define WMMA_BF16(Aop, Bop, Cop) \
  __builtin_amdgcn_wmma_f32_16x16x32_bf16(false, (Aop), false, (Bop), (short)0, (Cop), false, false)

// ---------------------------------------------------------------------------
// Implicit-GEMM conv. X: NHWC bf16, pixel stride CIp. W: [CO_pad][ks*ks*CIp] bf16.
// Yf: NCHW f32 (optional). Yb: NHWC bf16 stride ostr (optional).
// ---------------------------------------------------------------------------
__global__ __launch_bounds__(128)
void conv_wmma_kernel(const __bf16* __restrict__ X, const __bf16* __restrict__ Wt,
                      const float* __restrict__ bias, const float* __restrict__ resid,
                      float* __restrict__ Yf, __bf16* __restrict__ Yb,
                      int CI, int HI, int WI, int CO, int HO, int WO,
                      int ks, int stride, int pad, int CIp, int relu, int ostr) {
  const int tid  = threadIdx.x;
  const int lane = tid & 31;
  const int wave = tid >> 5;
  const int HWo  = HO * WO;
  const int P    = BN * HWo;
  const int ksq  = ks * ks;
  const int Kpad = ksq * CIp;
  const int nCoT = (CO + 15) >> 4;
  const int n0   = blockIdx.x * 64;
  const int coT  = blockIdx.y * 4 + wave;
  const int co0  = coT << 4;
  const bool act = (coT < nCoT);

  __shared__ __bf16 lds[4 * 16 * 32];

  // staging role: thread -> pixel n_st (0..63), 16-wide k half kh (0/16)
  const int n_st = tid >> 1;
  const int kh   = (tid & 1) << 4;
  const int p_st = n0 + n_st;
  const bool pv  = (p_st < P);
  int b_s = 0, y_s = 0, x_s = 0;
  if (pv) { b_s = p_st / HWo; int r = p_st - b_s * HWo; y_s = r / WO; x_s = r - y_s * WO; }
  __bf16* lws = lds + ((n_st >> 4) * 512 + (n_st & 15) * 32 + kh);

  const int m  = lane & 15;
  const int hf = lane >> 4;
  const __bf16* arow = Wt + (size_t)(co0 + m) * Kpad + hf * 16;

  v8f acc[4] = {};

  int ky = 0, kx = 0;
  for (int tap = 0; tap < ksq; ++tap) {
    const int yi = y_s * stride - pad + ky;
    const int xi = x_s * stride - pad + kx;
    const bool vt = pv && (unsigned)yi < (unsigned)HI && (unsigned)xi < (unsigned)WI;
    // NHWC pixel base (+kh channel offset); contiguous in channels.
    const __bf16* src = X + ((size_t)(b_s * HI + yi) * WI + xi) * CIp + kh;

    for (int ci0 = 0; ci0 < CIp; ci0 += 32) {
      if (vt) {
#if HAVE_ASYNC_LDS
        // global -> LDS, no VGPR round trip; IOFFSET applies to both sides.
        __builtin_amdgcn_global_load_async_to_lds_b128(
            (gas4_t)(src + ci0), (las4_t)lws, 0, 0);
        __builtin_amdgcn_global_load_async_to_lds_b128(
            (gas4_t)(src + ci0), (las4_t)lws, 16, 0);
#else
        *(v16bf*)lws = *(const v16bf*)(src + ci0);
#endif
      } else {
        v16bf z = {};
        *(v16bf*)lws = z;
      }
#if HAVE_ASYNC_LDS
      __builtin_amdgcn_s_wait_asynccnt(0);
#endif
      __syncthreads();
      if (act) {
        const int k0 = tap * CIp + ci0;
        if (k0 + 32 < Kpad) __builtin_prefetch(arow + k0 + 32, 0, 0);
        const __bf16* ap = arow + k0;
        const v8bf alo = *(const v8bf*)ap;            // k0+hf*16 + 0..7
        const v8bf ahi = *(const v8bf*)(ap + 8);      // k0+hf*16 + 8..15
        const v16bf A = __builtin_shufflevector(alo, ahi,
            0, 1, 2, 3, 4, 5, 6, 7, 8, 9, 10, 11, 12, 13, 14, 15);
        const __bf16* bp = lds + m * 32 + hf * 16;
        acc[0] = WMMA_BF16(A, *(const v16bf*)(bp        ), acc[0]);
        acc[1] = WMMA_BF16(A, *(const v16bf*)(bp +  512), acc[1]);
        acc[2] = WMMA_BF16(A, *(const v16bf*)(bp + 1024), acc[2]);
        acc[3] = WMMA_BF16(A, *(const v16bf*)(bp + 1536), acc[3]);
      }
      __syncthreads();
    }
    if (++kx == ks) { kx = 0; ++ky; }
  }

  // Epilogue (C/D layout: VGPR r -> M = r + 8*hf, N = lane&15)
  if (act) {
#pragma unroll
    for (int s = 0; s < 4; ++s) {
      const int p = n0 + s * 16 + m;
      if (p < P) {
        const int b  = p / HWo;
        const int r2 = p - b * HWo;
#pragma unroll
        for (int r = 0; r < 8; ++r) {
          const int co = co0 + hf * 8 + r;
          if (co < CO) {
            float v = acc[s][r];
            if (bias)  v += bias[co];
            if (resid) v += resid[(size_t)(b * CO + co) * HWo + r2];
            if (relu)  v = fmaxf(v, 0.f);
            if (Yf) Yf[(size_t)(b * CO + co) * HWo + r2] = v;          // NCHW f32
            if (Yb) Yb[(size_t)p * ostr + co] = (__bf16)v;             // NHWC bf16
          }
        }
      }
    }
  }
}

// ---------------------------------------------------------------------------
// InstanceNorm2d(affine=False) + ReLU: f32 NCHW in -> bf16 NHWC out (stride
// ostr, channel offset c_off).
// ---------------------------------------------------------------------------
__global__ __launch_bounds__(256)
void inorm_kernel(const float* __restrict__ X, __bf16* __restrict__ Y,
                  int C, int HWc, int relu, int c_off, int ostr) {
  const int bc = blockIdx.x;
  const int b = bc / C;
  const int c = bc - b * C;
  const float* src = X + (size_t)bc * HWc;
  float s = 0.f, s2 = 0.f;
  for (int i = threadIdx.x; i < HWc; i += 256) { float v = src[i]; s += v; s2 += v * v; }
  __shared__ float r1[256], r2[256];
  r1[threadIdx.x] = s; r2[threadIdx.x] = s2;
  __syncthreads();
  for (int off = 128; off > 0; off >>= 1) {
    if (threadIdx.x < off) { r1[threadIdx.x] += r1[threadIdx.x + off];
                             r2[threadIdx.x] += r2[threadIdx.x + off]; }
    __syncthreads();
  }
  const float inv = 1.f / (float)HWc;
  const float mu  = r1[0] * inv;
  const float var = fmaxf(r2[0] * inv - mu * mu, 0.f);
  const float rs  = rsqrtf(var + 1e-5f);
  __bf16* dst = Y + (size_t)b * HWc * ostr + c_off + c;
  for (int i = threadIdx.x; i < HWc; i += 256) {
    float v = (src[i] - mu) * rs;
    if (relu) v = fmaxf(v, 0.f);
    dst[(size_t)i * ostr] = (__bf16)v;
  }
}

// ---------------------------------------------------------------------------
// Correlation pyramid (NHWC bf16 inputs, stride 768 -> contiguous dots):
// out[b][dy*9+dx][y][x] = mean_c f1[b,y,x,c]*f2[b,y+dy-4,x+dx-4,c]
// ---------------------------------------------------------------------------
__global__ __launch_bounds__(256)
void corr_kernel(const __bf16* __restrict__ f1, const __bf16* __restrict__ f2,
                 float* __restrict__ out) {
  const int Cc = 768;
  int idx = blockIdx.x * 256 + threadIdx.x;
  if (idx >= BN * 9 * HWN) return;
  const int p = idx % HWN; const int r = idx / HWN;
  const int dy = r % 9;    const int b = r / 9;
  const int y = p / WN, x = p - y * WN;
  const int yy = y + dy - 4;
  float acc[9];
#pragma unroll
  for (int j = 0; j < 9; ++j) acc[j] = 0.f;
  if (yy >= 0 && yy < HN) {
    const __bf16* a = f1 + ((size_t)b * HWN + p) * Cc;
    const __bf16* base2 = f2 + ((size_t)b * HWN + yy * WN) * Cc;
    for (int j = 0; j < 9; ++j) {
      const int xx = x + j - 4;
      if (xx >= 0 && xx < WN) {
        const __bf16* bj = base2 + (size_t)xx * Cc;
        float s = 0.f;
        for (int c = 0; c < Cc; ++c) s += (float)a[c] * (float)bj[c];
        acc[j] = s;
      }
    }
  }
  float* o = out + ((size_t)b * 81 + dy * 9) * HWN + p;
#pragma unroll
  for (int j = 0; j < 9; ++j) o[(size_t)j * HWN] = acc[j] * (1.f / 768.f);
}

// ---------------------------------------------------------------------------
// corr_lookup: bilinear grid_sample on the (81, H*W) view (align_corners=True,
// border padding, reference's normalization). Writes gru_in ch [256..336].
// ---------------------------------------------------------------------------
__global__ __launch_bounds__(256)
void lookup_kernel(const float* __restrict__ corr, const float* __restrict__ disp,
                   __bf16* __restrict__ gin) {
  const int Kc = 81;
  int idx = blockIdx.x * 256 + threadIdx.x;
  if (idx >= BN * Kc * HWN) return;
  const int p = idx % HWN; const int r = idx / HWN;
  const int t = r % Kc;    const int b = r / Kc;
  const int y = p / WN, x = p - y * WN;
  const float gx = (float)x + disp[b * HWN + p] + (float)(t % 9) - 4.f;
  const float gy = (float)y + (float)(t / 9) - 4.f;
  float xs = gx * ((float)(HWN - 1) / (float)(WN - 1));
  float ys = gy * ((float)(Kc - 1) / (float)(HN - 1));
  xs = fminf(fmaxf(xs, 0.f), (float)(HWN - 1));
  ys = fminf(fmaxf(ys, 0.f), (float)(Kc - 1));
  const float xf = floorf(xs), yf = floorf(ys);
  const int x0 = (int)xf, y0 = (int)yf;
  const float wx = xs - xf, wy = ys - yf;
  const int x1 = min(x0 + 1, HWN - 1), y1 = min(y0 + 1, Kc - 1);
  const float* img = corr + (size_t)b * Kc * HWN;
  const float v00 = img[(size_t)y0 * HWN + x0], v01 = img[(size_t)y0 * HWN + x1];
  const float v10 = img[(size_t)y1 * HWN + x0], v11 = img[(size_t)y1 * HWN + x1];
  const float v = (1.f - wy) * ((1.f - wx) * v00 + wx * v01)
                +        wy  * ((1.f - wx) * v10 + wx * v11);
  gin[((size_t)b * HWN + p) * 352 + 256 + t] = (__bf16)v;
}

// GRU pointwise: z=sigmoid(g[0:128]), q=tanh(g[256:384]); h=(1-z)h+zq.
__global__ __launch_bounds__(256)
void gru_pw_kernel(const float* __restrict__ g, float* __restrict__ hidden,
                   __bf16* __restrict__ hb, __bf16* __restrict__ gin) {
  int idx = blockIdx.x * 256 + threadIdx.x;
  if (idx >= BN * 128 * HWN) return;
  const int p = idx % HWN; const int r = idx / HWN;
  const int c = r % 128;   const int b = r / 128;
  const float zv = g[((size_t)(b * 384) + c) * HWN + p];
  const float qv = g[((size_t)(b * 384) + 256 + c) * HWN + p];
  const float z = 1.f / (1.f + expf(-zv));
  const float q = tanhf(qv);
  const float hn = (1.f - z) * hidden[idx] + z * q;
  hidden[idx] = hn;
  hb[((size_t)b * HWN + p) * 128 + c] = (__bf16)hn;
  gin[((size_t)b * HWN + p) * 352 + c] = (__bf16)hn;
}

__global__ __launch_bounds__(256)
void disp_kernel(float* __restrict__ disp, const float* __restrict__ delta,
                 __bf16* __restrict__ gin) {
  int idx = blockIdx.x * 256 + threadIdx.x;
  if (idx >= BN * HWN) return;
  const int b = idx / HWN; const int p = idx - b * HWN;
  const float d = disp[idx] + delta[idx];
  disp[idx] = d;
  gin[((size_t)b * HWN + p) * 352 + 337] = (__bf16)d;
}

// Convex upsample: softmax over 9 mask logits, weighted 3x3 gather of 8*disp.
__global__ __launch_bounds__(256)
void upsample_kernel(const float* __restrict__ mlog, const float* __restrict__ disp,
                     float* __restrict__ out, int iter) {
  int idx = blockIdx.x * 256 + threadIdx.x;
  if (idx >= BN * HWN * 64) return;
  const int uv = idx & 63; const int rest = idx >> 6;
  const int p = rest % HWN; const int b = rest / HWN;
  const int u = uv >> 3, v = uv & 7;
  const int y = p / WN, x = p - y * WN;
  float e[9]; float mx = -1e30f;
#pragma unroll
  for (int k = 0; k < 9; ++k) {
    const float l = mlog[((size_t)(b * 576) + ((k * 8 + u) * 8 + v)) * HWN + p];
    e[k] = l; mx = fmaxf(mx, l);
  }
  float s = 0.f;
#pragma unroll
  for (int k = 0; k < 9; ++k) { e[k] = expf(e[k] - mx); s += e[k]; }
  float a = 0.f;
#pragma unroll
  for (int k = 0; k < 9; ++k) {
    const int ky = k / 3, kx = k - ky * 3;
    const int yy = y + ky - 1, xx = x + kx - 1;
    float dv = 0.f;
    if ((unsigned)yy < (unsigned)HN && (unsigned)xx < (unsigned)WN)
      dv = 8.f * disp[b * HWN + yy * WN + xx];
    a += e[k] * dv;
  }
  out[((size_t)(iter * 2 + b) * 448 + (y * 8 + u)) * 896 + (x * 8 + v)] = a / s;
}

// Pack OIHW f32 weights -> [CO_pad][tap][CI_pad] bf16, zero padded.
__global__ __launch_bounds__(256)
void pack_w_kernel(const float* __restrict__ w, __bf16* __restrict__ o,
                   int CO, int CI, int ks, int CIp, int n) {
  int i = blockIdx.x * 256 + threadIdx.x;
  if (i >= n) return;
  const int Kp = ks * ks * CIp;
  const int co = i / Kp;
  const int k  = i - co * Kp;
  const int tap = k / CIp;
  const int ci  = k - tap * CIp;
  const int ky = tap / ks, kx = tap - ky * ks;
  const float v = (co < CO && ci < CI)
      ? w[((size_t)(co * CI + ci) * ks + ky) * ks + kx] : 0.f;
  o[i] = (__bf16)v;
}

// f32 NCHW -> bf16 NHWC (pixel stride ostr), with scale.
__global__ __launch_bounds__(256)
void n2n_kernel(const float* __restrict__ src, __bf16* __restrict__ dst,
                int C, int HWc, int ostr, float sc, int n) {
  int idx = blockIdx.x * 256 + threadIdx.x;
  if (idx >= n) return;
  const int p = idx % HWc; const int r = idx / HWc;
  const int c = r % C;     const int b = r / C;
  dst[((size_t)b * HWc + p) * ostr + c] = (__bf16)(src[idx] * sc);
}

__global__ __launch_bounds__(256)
void zero_kernel(unsigned int* __restrict__ p, int n) {
  int i = blockIdx.x * 256 + threadIdx.x;
  if (i < n) p[i] = 0u;
}

// ---------------------------------------------------------------------------
extern "C" void kernel_launch(void* const* d_in, const int* in_sizes, int n_in,
                              void* d_out, int out_size, void* d_ws, size_t ws_size,
                              hipStream_t stream) {
  (void)in_sizes; (void)n_in; (void)out_size; (void)ws_size;

  const float* left  = (const float*)d_in[0];
  const float* fmap1 = (const float*)d_in[1];
  const float* fmap2 = (const float*)d_in[2];
  const float* ctxw[7];
  for (int i = 0; i < 7; ++i) ctxw[i] = (const float*)d_in[3 + i];
  const float* ad_w1 = (const float*)d_in[10];
  const float* ad_w2 = (const float*)d_in[11];
  const float* ad_b2 = (const float*)d_in[12];
  const float* gru_w = (const float*)d_in[13];
  const float* gru_b = (const float*)d_in[14];
  const float* dh_w1 = (const float*)d_in[15];
  const float* dh_b1 = (const float*)d_in[16];
  const float* dh_w2 = (const float*)d_in[17];
  const float* dh_b2 = (const float*)d_in[18];
  const float* up_w1 = (const float*)d_in[19];
  const float* up_b1 = (const float*)d_in[20];
  const float* up_w2 = (const float*)d_in[21];
  const float* up_b2 = (const float*)d_in[22];
  float* out = (float*)d_out;

  // ---- workspace carve-out ----
  char* cur = (char*)d_ws;
  auto alloc = [&](size_t elems, size_t esz) -> void* {
    void* r = (void*)cur;
    cur += (elems * esz + 255) & ~(size_t)255;
    return r;
  };
  const int ctxCI[7] = {3, 32, 32, 64, 64, 96, 96};
  const int ctxCO[7] = {32, 32, 64, 64, 96, 96, 128};
  int ctxCIp[7];
  __bf16* wb_ctx[7];
  for (int i = 0; i < 7; ++i) {
    ctxCIp[i] = (ctxCI[i] + 31) & ~31;
    wb_ctx[i] = (__bf16*)alloc((size_t)ctxCO[i] * 9 * ctxCIp[i], 2);
  }
  __bf16* wb_ad1 = (__bf16*)alloc((size_t)128 * 6912, 2);   // CIp=768, ks=3
  __bf16* wb_ad2 = (__bf16*)alloc((size_t)768 * 128, 2);    // CIp=128, ks=1
  __bf16* wb_gru = (__bf16*)alloc((size_t)384 * 3168, 2);   // CIp=352, ks=3
  __bf16* wb_dh1 = (__bf16*)alloc((size_t)256 * 1152, 2);   // CIp=128, ks=3
  __bf16* wb_dh2 = (__bf16*)alloc((size_t)16 * 256, 2);     // CIp=256, ks=1
  __bf16* wb_up1 = (__bf16*)alloc((size_t)256 * 1152, 2);
  __bf16* wb_up2 = (__bf16*)alloc((size_t)576 * 256, 2);
  // NHWC bf16 activations (pixel stride = consumer CI_pad)
  __bf16* img_bf  = (__bf16*)alloc((size_t)BN * 401408 * 32, 2);  // stride 32
  __bf16* bufA    = (__bf16*)alloc(6422528, 2);                   // ctx ping
  __bf16* bufB    = (__bf16*)alloc(6422528, 2);                   // ctx pong
  __bf16* fmap_bf = (__bf16*)alloc((size_t)BN * HWN * 768, 2);
  __bf16* f1a     = (__bf16*)alloc((size_t)BN * HWN * 768, 2);
  __bf16* f2a     = (__bf16*)alloc((size_t)BN * HWN * 768, 2);
  __bf16* h_ad    = (__bf16*)alloc((size_t)BN * HWN * 128, 2);
  __bf16* gru_in  = (__bf16*)alloc((size_t)BN * HWN * 352, 2);    // [h|ctx|corr|disp|pad]
  __bf16* hbf     = (__bf16*)alloc((size_t)BN * HWN * 128, 2);
  __bf16* mid_bf  = (__bf16*)alloc((size_t)BN * HWN * 256, 2);
  float* conv_f = (float*)alloc(7225344, 4);
  float* corr_p = (float*)alloc((size_t)BN * 81 * HWN, 4);
  float* hidden = (float*)alloc(1605632, 4);
  float* disp   = (float*)alloc(12544, 4);
  float* delta  = (float*)alloc(12544, 4);

  // ---- helpers ----
  auto conv = [&](const __bf16* X, const __bf16* Wt, const float* bias,
                  const float* resid, float* Yf, __bf16* Yb,
                  int CI, int HI, int WI, int CO, int HO, int WO,
                  int ks, int st, int pd, int CIp, int relu, int ostr) {
    const int P = BN * HO * WO;
    dim3 g(cdiv_h(P, 64), cdiv_h(cdiv_h(CO, 16), 4));
    conv_wmma_kernel<<<g, dim3(128), 0, stream>>>(X, Wt, bias, resid, Yf, Yb,
                                                  CI, HI, WI, CO, HO, WO,
                                                  ks, st, pd, CIp, relu, ostr);
  };
  auto inorm = [&](const float* X, __bf16* Y, int C, int HWc, int relu,
                   int c_off, int ostr) {
    inorm_kernel<<<BN * C, 256, 0, stream>>>(X, Y, C, HWc, relu, c_off, ostr);
  };
  auto pack = [&](const float* s, __bf16* d, int CO, int CI, int ks, int CIp) {
    const int COp = (CO + 15) & ~15;
    const int n = COp * ks * ks * CIp;
    pack_w_kernel<<<cdiv_h(n, 256), 256, 0, stream>>>(s, d, CO, CI, ks, CIp, n);
  };
  auto zero = [&](void* p, size_t words) {
    zero_kernel<<<cdiv_h((int)words, 256), 256, 0, stream>>>((unsigned int*)p, (int)words);
  };
  auto n2n = [&](const float* s, __bf16* d, int C, int HWc, int ostr, float sc) {
    const int n = BN * C * HWc;
    n2n_kernel<<<cdiv_h(n, 256), 256, 0, stream>>>(s, d, C, HWc, ostr, sc, n);
  };

  // ---- weight packing ----
  for (int i = 0; i < 7; ++i) pack(ctxw[i], wb_ctx[i], ctxCO[i], ctxCI[i], 3, ctxCIp[i]);
  pack(ad_w1, wb_ad1, 128, 768, 3, 768);
  pack(ad_w2, wb_ad2, 768, 128, 1, 128);
  pack(gru_w, wb_gru, 384, 338, 3, 352);
  pack(dh_w1, wb_dh1, 256, 128, 3, 128);
  pack(dh_w2, wb_dh2, 1, 256, 1, 256);
  pack(up_w1, wb_up1, 256, 128, 3, 128);
  pack(up_w2, wb_up2, 576, 256, 1, 256);

  // ---- state init (ws is poisoned; must re-zero every call) ----
  zero(hidden, 1605632);
  zero(disp, 12544);
  zero(gru_in, (size_t)BN * HWN * 352 * 2 / 4);
  zero(img_bf, (size_t)BN * 401408 * 32 * 2 / 4);   // pad channels must be 0-safe

  // ---- context encoder (img/255 NHWC(stride 32) -> 128ch into gru_in) ----
  n2n(left, img_bf, 3, 401408, 32, 1.f / 255.f);
  {
    const int st[7] = {2, 1, 2, 1, 2, 1, 1};
    int hi = 448, wi = 896;
    const __bf16* x = img_bf;
    __bf16* pp[2] = {bufA, bufB};
    for (int i = 0; i < 7; ++i) {
      const int ho = hi / st[i], wo = wi / st[i];
      conv(x, wb_ctx[i], nullptr, nullptr, conv_f, nullptr,
           ctxCI[i], hi, wi, ctxCO[i], ho, wo, 3, st[i], 1, ctxCIp[i], 0, 0);
      if (i < 6) {
        __bf16* dst = pp[i & 1];
        inorm(conv_f, dst, ctxCO[i], ho * wo, 1, 0, ctxCO[i]);  // ostr = next CIp
        x = dst;
      } else {
        inorm(conv_f, gru_in, 128, HWN, 1, 128, 352);           // ctx slot 128..255
      }
      hi = ho; wi = wo;
    }
  }

  // ---- feature adapter (both maps) ----
  auto adapter = [&](const float* fmap, __bf16* fa) {
    n2n(fmap, fmap_bf, 768, HWN, 768, 1.f);
    conv(fmap_bf, wb_ad1, nullptr, nullptr, conv_f, nullptr,
         768, HN, WN, 128, HN, WN, 3, 1, 1, 768, 0, 0);
    inorm(conv_f, h_ad, 128, HWN, 1, 0, 128);
    conv(h_ad, wb_ad2, ad_b2, fmap, nullptr, fa,     // fused bias + residual
         128, HN, WN, 768, HN, WN, 1, 1, 0, 128, 0, 768);
  };
  adapter(fmap1, f1a);
  adapter(fmap2, f2a);

  // ---- correlation pyramid ----
  corr_kernel<<<cdiv_h(BN * 9 * HWN, 256), 256, 0, stream>>>(f1a, f2a, corr_p);

  // ---- 12 RAFT iterations ----
  for (int it = 0; it < 12; ++it) {
    lookup_kernel<<<cdiv_h(BN * 81 * HWN, 256), 256, 0, stream>>>(corr_p, disp, gru_in);
    conv(gru_in, wb_gru, gru_b, nullptr, conv_f, nullptr,
         338, HN, WN, 384, HN, WN, 3, 1, 1, 352, 0, 0);
    gru_pw_kernel<<<cdiv_h(BN * 128 * HWN, 256), 256, 0, stream>>>(conv_f, hidden, hbf, gru_in);
    conv(hbf, wb_dh1, dh_b1, nullptr, nullptr, mid_bf,
         128, HN, WN, 256, HN, WN, 3, 1, 1, 128, 1, 256);
    conv(mid_bf, wb_dh2, dh_b2, nullptr, delta, nullptr,
         256, HN, WN, 1, HN, WN, 1, 1, 0, 256, 0, 0);
    disp_kernel<<<cdiv_h(BN * HWN, 256), 256, 0, stream>>>(disp, delta, gru_in);
    conv(hbf, wb_up1, up_b1, nullptr, nullptr, mid_bf,
         128, HN, WN, 256, HN, WN, 3, 1, 1, 128, 1, 256);
    conv(mid_bf, wb_up2, up_b2, nullptr, conv_f, nullptr,
         256, HN, WN, 576, HN, WN, 1, 1, 0, 256, 0, 0);
    upsample_kernel<<<cdiv_h(BN * HWN * 64, 256), 256, 0, stream>>>(conv_f, disp, out, it);
  }
}
